// LINDAAgent_38946763440974
// MI455X (gfx1250) — matrix operand order repaired and checked by reading, not compiled, not run on hardware
//
#include <hip/hip_runtime.h>
#include <math.h>

// ---------------------------------------------------------------------------
// Problem constants (match reference)
// ---------------------------------------------------------------------------
static constexpr int Bq    = 16384;
static constexpr int Nq    = 8;
static constexpr int RHq   = 64;
static constexpr int INq   = 128;
static constexpr int Rrows = Bq * Nq;          // 131072
static constexpr int Prows = Bq * Nq * Nq;     // 1048576
static constexpr float BN_EPS = 1e-5f;

// f16 weight image offsets (element counts) inside workspace
static constexpr int O_FC1 = 0;                 // 64*128
static constexpr int O_WIH = O_FC1 + 64 * 128;  // 192*64
static constexpr int O_WHH = O_WIH + 192 * 64;  // 192*64
static constexpr int O_AW1 = O_WHH + 192 * 64;  // 64*64
static constexpr int O_AW2 = O_AW1 + 64 * 64;   // 48*64
static constexpr int O_FC2 = O_AW2 + 48 * 64;   // 16*88
static constexpr int O_PO1 = O_FC2 + 16 * 88;   // 64*128
static constexpr int O_PO2 = O_PO1 + 64 * 128;  // 6*64
static constexpr int W16_TOTAL = O_PO2 + 6 * 64; // 49920 halves

typedef __attribute__((ext_vector_type(16))) _Float16 v16h;
typedef __attribute__((ext_vector_type(8)))  float    v8f;

// ---------------------------------------------------------------------------
// WMMA wrapper: D = A(16x32 f16) * B(32x16 f16) + C(16x16 f32)
// ---------------------------------------------------------------------------
static __device__ __forceinline__ v8f wmma_ff(v16h a, v16h b, v8f c) {
  return __builtin_amdgcn_wmma_f32_16x16x32_f16(
      /*neg_a=*/false, a, /*neg_b=*/false, b,
      /*c_mod=*/(short)0, c, /*reuse_a=*/false, /*reuse_b=*/false);
}

static __device__ __forceinline__ int lane_id() { return (int)(threadIdx.x & 31u); }

// ---------------------------------------------------------------------------
// Fragment loaders (CDNA5 wave32 16-bit layouts, CDNA5 ISA 7.12.2)
// ---------------------------------------------------------------------------
static __device__ __forceinline__ v16h afrag_f16(const _Float16* p) {
  int g = (lane_id() >> 4) & 1;
  p += g * 8;
  v16h a;
#pragma unroll
  for (int e = 0; e < 8; ++e) { a[e] = p[e]; a[e + 8] = p[e + 16]; }
  return a;
}

static __device__ __forceinline__ v16h afrag_f32(const float* p) {
  int g = (lane_id() >> 4) & 1;
  p += g * 8;
  v16h a;
#pragma unroll
  for (int e = 0; e < 8; ++e) {
    a[e]     = (_Float16)p[e];
    a[e + 8] = (_Float16)p[e + 16];
  }
  return a;
}

// A fragment with fused BN affine + ReLU (scale/shift indexed by column k)
static __device__ __forceinline__ v16h afrag_bn(const _Float16* p,
                                                const float* __restrict__ scale,
                                                const float* __restrict__ shift,
                                                int k0) {
  int g = (lane_id() >> 4) & 1;
  p  += g * 8;
  k0 += g * 8;
  v16h a;
#pragma unroll
  for (int e = 0; e < 8; ++e) {
    float x0 = (float)p[e]      * scale[k0 + e]      + shift[k0 + e];
    float x1 = (float)p[e + 16] * scale[k0 + e + 16] + shift[k0 + e + 16];
    a[e]     = (_Float16)fmaxf(x0, 0.f);
    a[e + 8] = (_Float16)fmaxf(x1, 0.f);
  }
  return a;
}

// B 32x16 f16 from pre-converted f16 weight W[nout][K] (row-major; works for
// global or LDS pointers). Column n of B == row n of W.
static __device__ __forceinline__ v16h bfrag_f16(const _Float16* w, int ld,
                                                 int n0, int k0, int nmax, int kmax) {
  int l = lane_id();
  int n = l & 15, g = l >> 4;
  int  nn  = n0 + n;
  bool nok = (nn < nmax);
  int  nc  = nok ? nn : 0;
  v16h b;
#pragma unroll
  for (int e = 0; e < 16; ++e) {
    int  kk  = k0 + g * 8 + (e & 7) + ((e >> 3) << 4);
    bool kok = (kk < kmax);
    int  kc  = kok ? kk : 0;
    _Float16 val = w[(size_t)nc * ld + kc];
    b[e] = (nok && kok) ? val : (_Float16)0.f;
  }
  return b;
}

// C fragment initialized with bias broadcast down columns
static __device__ __forceinline__ v8f cfrag_bias(const float* __restrict__ bias,
                                                 int n0, int nmax) {
  int n = lane_id() & 15;
  float bv = (n0 + n < nmax) ? bias[n0 + n] : 0.f;
  v8f c;
#pragma unroll
  for (int v = 0; v < 8; ++v) c[v] = bv;
  return c;
}

// ---------------------------------------------------------------------------
// Async DMA of a 16KB f16 weight tile into LDS (CDNA5 ASYNCcnt path).
// Each lane copies 16B per issue; 256 threads x 4 issues = 16KB.
// LDS byte address = low 32 bits of the generic shared pointer (flat aperture).
// ---------------------------------------------------------------------------
static __device__ __forceinline__ void stage_w_async(_Float16* lds_dst /*shared*/,
                                                     const _Float16* gsrc) {
  unsigned ldsbase = (unsigned)(unsigned long long)(const void*)lds_dst;
  int tid = (int)threadIdx.x;
#pragma unroll
  for (int it = 0; it < 4; ++it) {
    unsigned loff = ldsbase + (unsigned)((it * 256 + tid) * 16);
    unsigned long long gaddr =
        (unsigned long long)(const void*)((const char*)gsrc + (it * 256 + tid) * 16);
    asm volatile("global_load_async_to_lds_b128 %0, %1, off"
                 :: "v"(loff), "v"(gaddr) : "memory");
  }
  asm volatile("s_wait_asynccnt 0" ::: "memory");
  __syncthreads();
}

// ---------------------------------------------------------------------------
// K0: zero stats accumulators, seed KL with the constant -0.5 term
// ---------------------------------------------------------------------------
__global__ void k_init(float* __restrict__ stats, float* __restrict__ KL) {
  int t = blockIdx.x * blockDim.x + threadIdx.x;
  if (t < 512) stats[t] = 0.f;
  if (t < Bq)  KL[t] = -0.5f;
}

// ---------------------------------------------------------------------------
// K0b: one-shot fp32 -> f16 conversion of all weight matrices into ws
// ---------------------------------------------------------------------------
__global__ void k_cvt_all(const float* __restrict__ fc1, const float* __restrict__ wih,
                          const float* __restrict__ whh, const float* __restrict__ aw1,
                          const float* __restrict__ aw2, const float* __restrict__ fc2,
                          const float* __restrict__ po1, const float* __restrict__ po2,
                          _Float16* __restrict__ dst) {
  int t = blockIdx.x * blockDim.x + threadIdx.x;
  const float* s; int off;
  if      (t < O_WIH)      { s = fc1; off = t - O_FC1; }
  else if (t < O_WHH)      { s = wih; off = t - O_WIH; }
  else if (t < O_AW1)      { s = whh; off = t - O_WHH; }
  else if (t < O_AW2)      { s = aw1; off = t - O_AW1; }
  else if (t < O_FC2)      { s = aw2; off = t - O_AW2; }
  else if (t < O_PO1)      { s = fc2; off = t - O_FC2; }
  else if (t < O_PO2)      { s = po1; off = t - O_PO1; }
  else if (t < W16_TOTAL)  { s = po2; off = t - O_PO2; }
  else return;
  dst[t] = (_Float16)s[off];
}

// ---------------------------------------------------------------------------
// K1: x = relu(fc1(inputs));  h = GRUCell(x, hidden)  -> h_out(f32), h16(f16)
// ---------------------------------------------------------------------------
__global__ void __launch_bounds__(256)
k_fc1_gru(const float* __restrict__ inputs, const float* __restrict__ hidden,
          const _Float16* __restrict__ w16,
          const float* __restrict__ fc1_b, const float* __restrict__ b_ih,
          const float* __restrict__ b_hh,
          float* __restrict__ h_out, _Float16* __restrict__ h16) {
  __shared__ _Float16 lds_x[8][16 * 64];
  int wv = threadIdx.x >> 5, l = lane_id();
  int m = l & 15, g = l >> 4, n = l & 15;
  int r0 = (blockIdx.x * 8 + wv) * 16;
  _Float16* xw = lds_x[wv];
  const _Float16* fc1_w = w16 + O_FC1;
  const _Float16* w_ih  = w16 + O_WIH;
  const _Float16* w_hh  = w16 + O_WHH;

  // ---- fc1 + ReLU -> LDS (f16) ----
  v16h IA[4];
#pragma unroll
  for (int ks = 0; ks < 4; ++ks)
    IA[ks] = afrag_f32(inputs + (size_t)(r0 + m) * INq + ks * 32);
#pragma unroll
  for (int t = 0; t < 4; ++t) {
    v8f acc = cfrag_bias(fc1_b, t * 16, 64);
#pragma unroll
    for (int ks = 0; ks < 4; ++ks)
      acc = wmma_ff(IA[ks], bfrag_f16(fc1_w, INq, t * 16, ks * 32, 64, INq), acc);
#pragma unroll
    for (int v = 0; v < 8; ++v)
      xw[(v + 8 * g) * 64 + t * 16 + n] = (_Float16)fmaxf(acc[v], 0.f);
  }
  __syncthreads();

  // ---- GRU gates ----
  v16h XA[2], HA[2];
#pragma unroll
  for (int ks = 0; ks < 2; ++ks) {
    XA[ks] = afrag_f16(xw + m * 64 + ks * 32);
    HA[ks] = afrag_f32(hidden + (size_t)(r0 + m) * RHq + ks * 32);
  }
#pragma unroll
  for (int t = 0; t < 4; ++t) {
    v8f ir = cfrag_bias(b_ih, t * 16, 192);
    v8f iz = cfrag_bias(b_ih, 64 + t * 16, 192);
    v8f in_ = cfrag_bias(b_ih, 128 + t * 16, 192);
    v8f hr = cfrag_bias(b_hh, t * 16, 192);
    v8f hz = cfrag_bias(b_hh, 64 + t * 16, 192);
    v8f hn = cfrag_bias(b_hh, 128 + t * 16, 192);
#pragma unroll
    for (int ks = 0; ks < 2; ++ks) {
      ir  = wmma_ff(XA[ks], bfrag_f16(w_ih, 64, t * 16,       ks * 32, 192, 64), ir);
      iz  = wmma_ff(XA[ks], bfrag_f16(w_ih, 64, 64 + t * 16,  ks * 32, 192, 64), iz);
      in_ = wmma_ff(XA[ks], bfrag_f16(w_ih, 64, 128 + t * 16, ks * 32, 192, 64), in_);
      hr  = wmma_ff(HA[ks], bfrag_f16(w_hh, 64, t * 16,       ks * 32, 192, 64), hr);
      hz  = wmma_ff(HA[ks], bfrag_f16(w_hh, 64, 64 + t * 16,  ks * 32, 192, 64), hz);
      hn  = wmma_ff(HA[ks], bfrag_f16(w_hh, 64, 128 + t * 16, ks * 32, 192, 64), hn);
    }
#pragma unroll
    for (int v = 0; v < 8; ++v) {
      int row = r0 + v + 8 * g;
      int col = t * 16 + n;
      float hold = hidden[(size_t)row * RHq + col];
      float rg = 1.f / (1.f + __expf(-(ir[v] + hr[v])));
      float zg = 1.f / (1.f + __expf(-(iz[v] + hz[v])));
      float ng = tanhf(in_[v] + rg * hn[v]);
      float hv = (1.f - zg) * ng + zg * hold;
      h_out[(size_t)row * RHq + col] = hv;
      h16[(size_t)row * RHq + col]   = (_Float16)hv;
    }
  }
}

// ---------------------------------------------------------------------------
// K2: t1 = h @ aw1_w.T + aw1_b  (pre-BN, f16) + per-column sum/sumsq stats
// ---------------------------------------------------------------------------
__global__ void __launch_bounds__(256)
k_aw1(const _Float16* __restrict__ h16, const _Float16* __restrict__ w16,
      const float* __restrict__ aw1_b, _Float16* __restrict__ t1,
      float* __restrict__ sumC, float* __restrict__ sqC) {
  __shared__ float csum[64], csq[64];
  if (threadIdx.x < 64) { csum[threadIdx.x] = 0.f; csq[threadIdx.x] = 0.f; }
  __syncthreads();
  int wv = threadIdx.x >> 5, l = lane_id();
  int m = l & 15, g = l >> 4, n = l & 15;
  int r0 = (blockIdx.x * 8 + wv) * 16;
  const _Float16* aw1_w = w16 + O_AW1;
  v16h HA[2];
#pragma unroll
  for (int ks = 0; ks < 2; ++ks)
    HA[ks] = afrag_f16(h16 + (size_t)(r0 + m) * RHq + ks * 32);
#pragma unroll
  for (int t = 0; t < 4; ++t) {
    v8f acc = cfrag_bias(aw1_b, t * 16, 64);
#pragma unroll
    for (int ks = 0; ks < 2; ++ks)
      acc = wmma_ff(HA[ks], bfrag_f16(aw1_w, 64, t * 16, ks * 32, 64, 64), acc);
    float s1 = 0.f, s2 = 0.f;
#pragma unroll
    for (int v = 0; v < 8; ++v) {
      t1[(size_t)(r0 + v + 8 * g) * 64 + t * 16 + n] = (_Float16)acc[v];
      s1 += acc[v]; s2 += acc[v] * acc[v];
    }
    atomicAdd(&csum[t * 16 + n], s1);
    atomicAdd(&csq[t * 16 + n], s2);
  }
  __syncthreads();
  if (threadIdx.x < 64) {
    atomicAdd(&sumC[threadIdx.x], csum[threadIdx.x]);
    atomicAdd(&sqC[threadIdx.x],  csq[threadIdx.x]);
  }
}

// ---------------------------------------------------------------------------
// K3: finalize BN: scale = g*rsqrt(var+eps), shift = b - mean*scale
// ---------------------------------------------------------------------------
__global__ void k_bnfin(const float* __restrict__ sum, const float* __restrict__ sq,
                        float invN, const float* __restrict__ gamma,
                        const float* __restrict__ beta,
                        float* __restrict__ scale, float* __restrict__ shift) {
  int c = threadIdx.x;  // 64 threads
  float mean = sum[c] * invN;
  float var  = sq[c] * invN - mean * mean;
  float sc   = gamma[c] * rsqrtf(var + BN_EPS);
  scale[c] = sc;
  shift[c] = beta[c] - mean * sc;
}

// ---------------------------------------------------------------------------
// K4: aw = relu(bn(t1)) @ aw2_w.T + b -> mu/sr to ws; c = mu + exp(sr)*eps;
//     q = [h, c] @ fc2_w.T + fc2_b
// ---------------------------------------------------------------------------
__global__ void __launch_bounds__(256)
k_aw2_fc2(const _Float16* __restrict__ h16, const _Float16* __restrict__ t1,
          const _Float16* __restrict__ w16,
          const float* __restrict__ awScale, const float* __restrict__ awShift,
          const float* __restrict__ aw2_b, const float* __restrict__ eps,
          const float* __restrict__ fc2_b,
          float* __restrict__ mu_ws, float* __restrict__ sr_ws,
          float* __restrict__ q_out) {
  __shared__ float     awt[8][16 * 48];
  __shared__ _Float16  ctl[8][16 * 32];
  int wv = threadIdx.x >> 5, l = lane_id();
  int m = l & 15, g = l >> 4, n = l & 15;
  int r0 = (blockIdx.x * 8 + wv) * 16;
  float*    aw_w = awt[wv];
  _Float16* ct_w = ctl[wv];
  const _Float16* aw2_w = w16 + O_AW2;
  const _Float16* fc2_w = w16 + O_FC2;

  v16h YA[2];
#pragma unroll
  for (int ks = 0; ks < 2; ++ks)
    YA[ks] = afrag_bn(t1 + (size_t)(r0 + m) * 64 + ks * 32, awScale, awShift, ks * 32);
#pragma unroll
  for (int t = 0; t < 3; ++t) {
    v8f acc = cfrag_bias(aw2_b, t * 16, 48);
#pragma unroll
    for (int ks = 0; ks < 2; ++ks)
      acc = wmma_ff(YA[ks], bfrag_f16(aw2_w, 64, t * 16, ks * 32, 48, 64), acc);
#pragma unroll
    for (int v = 0; v < 8; ++v) {
      int row = v + 8 * g, c48 = t * 16 + n;
      aw_w[row * 48 + c48] = acc[v];
      int gr = r0 + row;
      if (c48 < 24) mu_ws[(size_t)gr * 24 + c48]      = acc[v];
      else          sr_ws[(size_t)gr * 24 + c48 - 24] = acc[v];
    }
  }
  __syncthreads();
  // c = mu + exp(sr) * eps  -> LDS f16 tile (24 cols + 8 zero pad)
  for (int idx = l; idx < 16 * 24; idx += 32) {
    int row = idx / 24, d = idx % 24;
    float muv = aw_w[row * 48 + d];
    float srv = aw_w[row * 48 + 24 + d];
    float ev  = eps[(size_t)(r0 + row) * 24 + d];
    ct_w[row * 32 + d] = (_Float16)(muv + __expf(srv) * ev);
  }
  for (int idx = l; idx < 16 * 8; idx += 32)
    ct_w[(idx / 8) * 32 + 24 + (idx % 8)] = (_Float16)0.f;
  __syncthreads();
  // fc2: K = 88 = 64 (h) + 24 (c, padded to 32)
  v8f acc = cfrag_bias(fc2_b, 0, 16);
  v16h HA0 = afrag_f16(h16 + (size_t)(r0 + m) * 64 + 0);
  v16h HA1 = afrag_f16(h16 + (size_t)(r0 + m) * 64 + 32);
  v16h CA  = afrag_f16(ct_w + m * 32);
  acc = wmma_ff(HA0, bfrag_f16(fc2_w, 88, 0, 0,  16, 88), acc);
  acc = wmma_ff(HA1, bfrag_f16(fc2_w, 88, 0, 32, 16, 88), acc);
  acc = wmma_ff(CA,  bfrag_f16(fc2_w, 88, 0, 64, 16, 88), acc);
#pragma unroll
  for (int v = 0; v < 8; ++v)
    q_out[(size_t)(r0 + v + 8 * g) * 16 + n] = acc[v];
}

// ---------------------------------------------------------------------------
// Pair-row A fragments: row r -> (b,i,j); K<64 from h[b*8+j], K>=64 from h[b*8+i]
// ---------------------------------------------------------------------------
static __device__ __forceinline__ void pair_afrags(const _Float16* __restrict__ h16,
                                                   int r0, v16h A[4]) {
  int m = lane_id() & 15;
  int row = r0 + m;
  int b = row >> 6, i = (row >> 3) & 7, j = row & 7;
  const _Float16* pj = h16 + (size_t)(b * 8 + j) * 64;
  const _Float16* pi = h16 + (size_t)(b * 8 + i) * 64;
  A[0] = afrag_f16(pj);       A[1] = afrag_f16(pj + 32);
  A[2] = afrag_f16(pi);       A[3] = afrag_f16(pi + 32);
}

// ---------------------------------------------------------------------------
// K5: po1 pre-BN stats pass; po1_w staged to LDS via async DMA
// ---------------------------------------------------------------------------
__global__ void __launch_bounds__(256)
k_po1_stats(const _Float16* __restrict__ h16, const _Float16* __restrict__ w16,
            const float* __restrict__ po1_b,
            float* __restrict__ sumC, float* __restrict__ sqC) {
  __shared__ float csum[64], csq[64];
  __shared__ alignas(16) _Float16 wlds[64 * 128];
  if (threadIdx.x < 64) { csum[threadIdx.x] = 0.f; csq[threadIdx.x] = 0.f; }
  stage_w_async(wlds, w16 + O_PO1);   // includes __syncthreads()

  int wv = threadIdx.x >> 5, l = lane_id();
  int n = l & 15;
  int r0 = (blockIdx.x * 8 + wv) * 16;
  v16h A[4];
  pair_afrags(h16, r0, A);
#pragma unroll
  for (int t = 0; t < 4; ++t) {
    v8f acc = cfrag_bias(po1_b, t * 16, 64);
#pragma unroll
    for (int ks = 0; ks < 4; ++ks)
      acc = wmma_ff(A[ks], bfrag_f16(wlds, 128, t * 16, ks * 32, 64, 128), acc);
    float s1 = 0.f, s2 = 0.f;
#pragma unroll
    for (int v = 0; v < 8; ++v) { s1 += acc[v]; s2 += acc[v] * acc[v]; }
    atomicAdd(&csum[t * 16 + n], s1);
    atomicAdd(&csq[t * 16 + n], s2);
  }
  __syncthreads();
  if (threadIdx.x < 64) {
    atomicAdd(&sumC[threadIdx.x], csum[threadIdx.x]);
    atomicAdd(&sqC[threadIdx.x],  csq[threadIdx.x]);
  }
}

// ---------------------------------------------------------------------------
// K7: recompute po1 (LDS-staged weights), BN+ReLU, po2, accumulate KL
// ---------------------------------------------------------------------------
__global__ void __launch_bounds__(256)
k_po2_kl(const _Float16* __restrict__ h16, const _Float16* __restrict__ w16,
         const float* __restrict__ po1_b, const float* __restrict__ poScale,
         const float* __restrict__ poShift, const float* __restrict__ po2_b,
         const float* __restrict__ mu_ws, const float* __restrict__ sr_ws,
         float* __restrict__ KL) {
  __shared__ _Float16 ylds[8][16 * 64];
  __shared__ alignas(16) _Float16 wlds[64 * 128];
  stage_w_async(wlds, w16 + O_PO1);   // includes __syncthreads()

  int wv = threadIdx.x >> 5, l = lane_id();
  int m = l & 15, g = l >> 4, n = l & 15;
  int r0 = (blockIdx.x * 8 + wv) * 16;
  _Float16* yw = ylds[wv];
  const _Float16* po2_w = w16 + O_PO2;

  v16h A[4];
  pair_afrags(h16, r0, A);
#pragma unroll
  for (int t = 0; t < 4; ++t) {
    v8f acc = cfrag_bias(po1_b, t * 16, 64);
#pragma unroll
    for (int ks = 0; ks < 4; ++ks)
      acc = wmma_ff(A[ks], bfrag_f16(wlds, 128, t * 16, ks * 32, 64, 128), acc);
#pragma unroll
    for (int v = 0; v < 8; ++v) {
      int col = t * 16 + n;
      float y = fmaxf(acc[v] * poScale[col] + poShift[col], 0.f);
      yw[(v + 8 * g) * 64 + col] = (_Float16)y;
    }
  }
  __syncthreads();
  // po2: 6 output cols (padded to 16), K = 64
  v16h Y0 = afrag_f16(yw + m * 64);
  v16h Y1 = afrag_f16(yw + m * 64 + 32);
  v8f acc = cfrag_bias(po2_b, 0, 6);
  acc = wmma_ff(Y0, bfrag_f16(po2_w, 64, 0, 0,  6, 64), acc);
  acc = wmma_ff(Y1, bfrag_f16(po2_w, 64, 0, 32, 6, 64), acc);
  // KL: rows j=v (mu1) pair with j=v+4 (log-sigma1) within the same lane
  int b  = r0 >> 6;
  int i0 = (r0 >> 3) & 7;   // even; lane group g owns i0+g
  if (n < 6) {
    int gr = b * 8 + i0 + g;
    float s = 0.f;
#pragma unroll
    for (int v = 0; v < 4; ++v) {
      int dd = v * 6 + n;
      float muv = mu_ws[(size_t)gr * 24 + dd];
      float srv = sr_ws[(size_t)gr * 24 + dd];
      float mu1 = acc[v];
      float tt  = acc[v + 4];
      float dmu = muv - mu1;
      s += (tt - srv) + (__expf(2.f * srv) + dmu * dmu) * 0.5f * __expf(-2.f * tt);
    }
    atomicAdd(&KL[b], s * (1.f / 192.f));
  }
}

// ---------------------------------------------------------------------------
// Host launcher
// ---------------------------------------------------------------------------
extern "C" void kernel_launch(void* const* d_in, const int* in_sizes, int n_in,
                              void* d_out, int out_size, void* d_ws, size_t ws_size,
                              hipStream_t stream) {
  (void)in_sizes; (void)n_in; (void)out_size; (void)ws_size;
  const float* inputs  = (const float*)d_in[0];
  const float* hidden  = (const float*)d_in[1];
  const float* eps     = (const float*)d_in[2];
  const float* fc1_w   = (const float*)d_in[3];
  const float* fc1_b   = (const float*)d_in[4];
  const float* gw_ih   = (const float*)d_in[5];
  const float* gw_hh   = (const float*)d_in[6];
  const float* gb_ih   = (const float*)d_in[7];
  const float* gb_hh   = (const float*)d_in[8];
  const float* aw1_w   = (const float*)d_in[9];
  const float* aw1_b   = (const float*)d_in[10];
  const float* aw_bn_g = (const float*)d_in[11];
  const float* aw_bn_b = (const float*)d_in[12];
  const float* aw2_w   = (const float*)d_in[13];
  const float* aw2_b   = (const float*)d_in[14];
  const float* po1_w   = (const float*)d_in[15];
  const float* po1_b   = (const float*)d_in[16];
  const float* po_bn_g = (const float*)d_in[17];
  const float* po_bn_b = (const float*)d_in[18];
  const float* po2_w   = (const float*)d_in[19];
  const float* po2_b   = (const float*)d_in[20];
  const float* fc2_w   = (const float*)d_in[21];
  const float* fc2_b   = (const float*)d_in[22];

  // d_out: q [R,16] | h [R,64] | KL [B]
  float* q_out = (float*)d_out;
  float* h_out = q_out + (size_t)Rrows * 16;
  float* KL    = h_out + (size_t)Rrows * 64;

  // workspace layout
  char* ws = (char*)d_ws;
  float* stats = (float*)ws;                   // 512 floats
  float* awSum = stats,        *awSq = stats + 64;
  float* poSum = stats + 128,  *poSq = stats + 192;
  float* awScale = stats + 256, *awShift = stats + 320;
  float* poScale = stats + 384, *poShift = stats + 448;
  _Float16* h16 = (_Float16*)(ws + 4096);
  _Float16* t1  = h16 + (size_t)Rrows * 64;
  float* mu_ws  = (float*)(t1 + (size_t)Rrows * 64);
  float* sr_ws  = mu_ws + (size_t)Rrows * 24;
  _Float16* w16 = (_Float16*)(sr_ws + (size_t)Rrows * 24);   // 49920 halves

  dim3 blk(256);
  k_init<<<dim3((Bq + 255) / 256), blk, 0, stream>>>(stats, KL);
  k_cvt_all<<<dim3((W16_TOTAL + 255) / 256), blk, 0, stream>>>(
      fc1_w, gw_ih, gw_hh, aw1_w, aw2_w, fc2_w, po1_w, po2_w, w16);
  k_fc1_gru<<<dim3(Rrows / 128), blk, 0, stream>>>(
      inputs, hidden, w16, fc1_b, gb_ih, gb_hh, h_out, h16);
  k_aw1<<<dim3(Rrows / 128), blk, 0, stream>>>(h16, w16, aw1_b, t1, awSum, awSq);
  k_bnfin<<<dim3(1), dim3(64), 0, stream>>>(awSum, awSq, 1.f / (float)Rrows,
                                            aw_bn_g, aw_bn_b, awScale, awShift);
  k_aw2_fc2<<<dim3(Rrows / 128), blk, 0, stream>>>(
      h16, t1, w16, awScale, awShift, aw2_b, eps, fc2_b, mu_ws, sr_ws, q_out);
  k_po1_stats<<<dim3(Prows / 128), blk, 0, stream>>>(h16, w16, po1_b, poSum, poSq);
  k_bnfin<<<dim3(1), dim3(64), 0, stream>>>(poSum, poSq, 1.f / (float)Prows,
                                            po_bn_g, po_bn_b, poScale, poShift);
  k_po2_kl<<<dim3(Prows / 128), blk, 0, stream>>>(
      h16, w16, po1_b, poScale, poShift, po2_b, mu_ws, sr_ws, KL);
}
